// SingleStreamBlock_32212254720695
// MI455X (gfx1250) — compile-verified, matrix operand order checked by reading, not run on hardware
//
#include <hip/hip_runtime.h>
#include <hip/hip_bf16.h>

// ---------------------------------------------------------------------------
// SingleStreamBlock for MI455X (gfx1250, wave32, WMMA f16->f32)
// B=1, L=2048, H=3072, NH=24, D=128, M=12288
// ---------------------------------------------------------------------------

typedef __attribute__((ext_vector_type(16))) _Float16 v16h;
typedef __attribute__((ext_vector_type(8)))  _Float16 v8h;
typedef __attribute__((ext_vector_type(8)))  float    v8f;
typedef __attribute__((ext_vector_type(4)))  unsigned int u32x4;
typedef __attribute__((ext_vector_type(8)))  int      i32x8;
typedef __attribute__((ext_vector_type(4)))  int      i32x4;
// exact pointee type of the async-to-LDS builtin params (per clang diagnostic)
typedef int async_i4 __attribute__((__vector_size__(4 * sizeof(int))));

#define WMMA_F16(a, b, c) \
  __builtin_amdgcn_wmma_f32_16x16x32_f16(false, (a), false, (b), (short)0, (c), false, false)

// CDNA5-specific data movement, feature-guarded so absence degrades gracefully.
#if defined(__gfx1250__)
#define HAS_ASYNC_LDS __has_builtin(__builtin_amdgcn_global_load_async_to_lds_b128)
#define HAS_TDM       __has_builtin(__builtin_amdgcn_tensor_load_to_lds)
#else
#define HAS_ASYNC_LDS 0
#define HAS_TDM 0
#endif

static constexpr int Lq = 2048;
static constexpr int Hd = 3072;
static constexpr int NH = 24;
static constexpr int Dh = 128;
static constexpr int Mm = 12288;
static constexpr int BIGK = Hd + Mm;   // 15360

__device__ inline void wait_async_all() {
#if HAS_ASYNC_LDS
#if __has_builtin(__builtin_amdgcn_s_wait_asynccnt)
  __builtin_amdgcn_s_wait_asynccnt(0);
#else
  asm volatile("s_wait_asynccnt 0x0" ::: "memory");
#endif
#endif
}

__device__ inline void wait_tensor_all() {
#if HAS_TDM
#if __has_builtin(__builtin_amdgcn_s_wait_tensorcnt)
  __builtin_amdgcn_s_wait_tensorcnt(0);
#else
  asm volatile("s_wait_tensorcnt 0x0" ::: "memory");
#endif
#endif
}

// --------------------------------------------------------------------------
// Fragment loader: 16-bit A/B matrix, 16x32 (MxK) per ISA layout:
//   lanes 0-15 hold row (lane&15), K = {ks..ks+7, ks+16..ks+23}
//   lanes 16-31 hold row (lane&15), K = {ks+8..ks+15, ks+24..ks+31}
// ld must be a multiple of 8 halfs so the 16B LDS loads stay aligned.
// --------------------------------------------------------------------------
__device__ inline v16h frag16x32(const _Float16* rowbase, int ld, int ks, int lane) {
  const _Float16* p = rowbase + (lane & 15) * ld + ks + ((lane >> 4) << 3);
  v8h lo = *(const v8h*)p;
  v8h hi = *(const v8h*)(p + 16);
  v16h o;
#pragma unroll
  for (int i = 0; i < 8; ++i) { o[i] = lo[i]; o[i + 8] = hi[i]; }
  return o;
}

// --------------------------------------------------------------------------
// K0: silu(vec)
// --------------------------------------------------------------------------
__global__ void silu_vec_kernel(const float* __restrict__ vec, float* __restrict__ sv) {
  int i = blockIdx.x * 256 + threadIdx.x;
  if (i < Hd) {
    float v = vec[i];
    sv[i] = v / (1.f + __expf(-v));
  }
}

// --------------------------------------------------------------------------
// K1: mod = silu(vec) @ mod_w^T + mod_b   (9216 x 3072 GEMV, memory bound)
// --------------------------------------------------------------------------
__global__ __launch_bounds__(256) void mod_gemv_kernel(
    const float* __restrict__ sv, const float* __restrict__ mw,
    const float* __restrict__ mb, float* __restrict__ mod) {
  int o = blockIdx.x * 256 + threadIdx.x;
  if (o >= 3 * Hd) return;
  const float4* wr = (const float4*)(mw + (size_t)o * Hd);
  const float4* s4 = (const float4*)sv;
  float acc = 0.f;
  for (int i = 0; i < Hd / 4; ++i) {
    float4 w = wr[i], s = s4[i];
    acc += w.x * s.x + w.y * s.y + w.z * s.z + w.w * s.w;
  }
  mod[o] = acc + mb[o];
}

// --------------------------------------------------------------------------
// K2: LayerNorm (no affine) + modulation -> x_mod (f16)
// --------------------------------------------------------------------------
__global__ __launch_bounds__(256) void ln_mod_kernel(
    const float* __restrict__ x, const float* __restrict__ mod,
    _Float16* __restrict__ xmod) {
  int row = blockIdx.x;
  const float* xr = x + (size_t)row * Hd;
  float vals[12];
  float s = 0.f, ss = 0.f;
#pragma unroll
  for (int t = 0; t < 12; ++t) {
    float v = xr[threadIdx.x + t * 256];
    vals[t] = v; s += v; ss += v * v;
  }
  __shared__ float red1[8], red2[8];
  int lane = threadIdx.x & 31, w = threadIdx.x >> 5;
#pragma unroll
  for (int off = 16; off > 0; off >>= 1) {
    s  += __shfl_down(s, off, 32);
    ss += __shfl_down(ss, off, 32);
  }
  if (lane == 0) { red1[w] = s; red2[w] = ss; }
  __syncthreads();
  if (w == 0) {
    s  = (lane < 8) ? red1[lane] : 0.f;
    ss = (lane < 8) ? red2[lane] : 0.f;
#pragma unroll
    for (int off = 4; off > 0; off >>= 1) {
      s  += __shfl_down(s, off, 32);
      ss += __shfl_down(ss, off, 32);
    }
    if (lane == 0) { red1[0] = s; red2[0] = ss; }
  }
  __syncthreads();
  float mu   = red1[0] * (1.f / Hd);
  float var  = red2[0] * (1.f / Hd) - mu * mu;
  float rstd = rsqrtf(var + 1e-6f);
#pragma unroll
  for (int t = 0; t < 12; ++t) {
    int c = threadIdx.x + t * 256;
    float xn = (vals[t] - mu) * rstd;
    float v = (1.f + mod[Hd + c]) * xn + mod[c];
    xmod[(size_t)row * Hd + c] = (_Float16)v;
  }
}

// --------------------------------------------------------------------------
// Tile staging helpers for the WMMA GEMM (double buffered)
// --------------------------------------------------------------------------
__device__ inline void stage_a_tile(const _Float16* __restrict__ A, int lda,
                                    int m0, int kb, _Float16* dstbuf, int tid) {
#pragma unroll
  for (int t = 0; t < 4; ++t) {
    int chunk = tid + t * 256;
    int r = chunk >> 3, c0 = (chunk & 7) * 8;
    const _Float16* src = A + (size_t)(m0 + r) * lda + kb + c0;
    _Float16* dst = dstbuf + r * 72 + c0;
#if HAS_ASYNC_LDS
    __builtin_amdgcn_global_load_async_to_lds_b128(
        (__attribute__((address_space(1))) async_i4*)(unsigned long long)src,
        (__attribute__((address_space(3))) async_i4*)(unsigned long long)dst, 0, 0);
#else
    *(v8h*)dst = *(const v8h*)src;
#endif
  }
}

__device__ inline void stage_w_tile(const float* __restrict__ W, int ldw,
                                    int n0, int kb, _Float16* dstbuf, int tid) {
#pragma unroll
  for (int t = 0; t < 4; ++t) {
    int chunk = tid + t * 256;
    int r = chunk >> 4, c0 = (chunk & 15) * 4;
    float4 wv = *(const float4*)(W + (size_t)(n0 + r) * ldw + kb + c0);
    _Float16* dst = dstbuf + r * 72 + c0;
    dst[0] = (_Float16)wv.x; dst[1] = (_Float16)wv.y;
    dst[2] = (_Float16)wv.z; dst[3] = (_Float16)wv.w;
  }
}

// --------------------------------------------------------------------------
// Generic WMMA GEMM: C[M,N] = A[M,K](f16) @ W[N,K]^T(f32->f16) + epilogue
// Block tile 128x64, 8 waves, each wave 32x32 (2x2 WMMA tiles), KB=64.
// Double-buffered LDS; A tile staged with async-to-LDS when available.
// --------------------------------------------------------------------------
enum { EP_QKV = 0, EP_GELU = 1, EP_OUT = 2 };

template <int EP>
__global__ __launch_bounds__(256) void wmma_gemm_kernel(
    const _Float16* __restrict__ A, int lda,
    const float* __restrict__ W, int ldw,
    const float* __restrict__ bias, int K,
    _Float16* __restrict__ Ch, int ldc, int ccol0,
    float* __restrict__ Cf,
    const float* __restrict__ xres,
    const float* __restrict__ mod) {
  constexpr int LDS_A = 72;  // 64 + 8 pad (keeps 16B alignment)
  constexpr int LDS_W = 72;
  __shared__ _Float16 sA[2][128 * LDS_A];
  __shared__ _Float16 sW[2][64 * LDS_W];

  int m0 = blockIdx.y * 128;
  int n0 = blockIdx.x * 64;
  int tid = threadIdx.x, lane = tid & 31, w = tid >> 5;
  int wr = (w & 3) * 32;   // M offset within 128
  int wc = (w >> 2) * 32;  // N offset within 64

  v8f acc[2][2] = {};

  stage_a_tile(A, lda, m0, 0, sA[0], tid);
  stage_w_tile(W, ldw, n0, 0, sW[0], tid);

  for (int kb = 0; kb < K; kb += 64) {
    int cur = (kb >> 6) & 1;
    wait_async_all();        // this wave's pending async copies into sA[cur]
    __syncthreads();         // everyone's staging of buffer `cur` visible
    if (kb + 64 < K) {
      __builtin_prefetch(W + (size_t)(n0 + (tid >> 2)) * ldw + kb + 64 + (tid & 3) * 16, 0, 1);
      stage_a_tile(A, lda, m0, kb + 64, sA[cur ^ 1], tid);
      stage_w_tile(W, ldw, n0, kb + 64, sW[cur ^ 1], tid);
    }
#pragma unroll
    for (int ks = 0; ks < 64; ks += 32) {
      v16h a0 = frag16x32(sA[cur] + wr * LDS_A,        LDS_A, ks, lane);
      v16h a1 = frag16x32(sA[cur] + (wr + 16) * LDS_A, LDS_A, ks, lane);
      v16h b0 = frag16x32(sW[cur] + wc * LDS_W,        LDS_W, ks, lane);
      v16h b1 = frag16x32(sW[cur] + (wc + 16) * LDS_W, LDS_W, ks, lane);
      acc[0][0] = WMMA_F16(a0, b0, acc[0][0]);
      acc[0][1] = WMMA_F16(a0, b1, acc[0][1]);
      acc[1][0] = WMMA_F16(a1, b0, acc[1][0]);
      acc[1][1] = WMMA_F16(a1, b1, acc[1][1]);
    }
  }

  // epilogue: C/D layout -> VGPR i: lanes 0-15 M=i, lanes 16-31 M=i+8; N=lane&15
  int hi8 = (lane >> 4) << 3;
  int cn  = lane & 15;
#pragma unroll
  for (int ti = 0; ti < 2; ++ti) {
#pragma unroll
    for (int tj = 0; tj < 2; ++tj) {
#pragma unroll
      for (int i = 0; i < 8; ++i) {
        int row = m0 + wr + ti * 16 + i + hi8;
        int col = n0 + wc + tj * 16 + cn;
        float vacc = acc[ti][tj][i];
        if (EP == EP_QKV) {
          Ch[(size_t)row * ldc + ccol0 + col] = (_Float16)(vacc + bias[col]);
        } else if (EP == EP_GELU) {
          float xg = vacc + bias[col];
          float g = 0.5f * xg *
                    (1.f + tanhf(0.7978845608028654f * (xg + 0.044715f * xg * xg * xg)));
          Ch[(size_t)row * ldc + ccol0 + col] = (_Float16)g;
        } else {  // EP_OUT: out = x + gate * (acc + 2*b2)
          float val = vacc + 2.f * bias[col];
          Cf[(size_t)row * ldc + col] =
              xres[(size_t)row * ldc + col] + mod[2 * Hd + col] * val;
        }
      }
    }
  }
}

// --------------------------------------------------------------------------
// K4: per-(l,head) RMSNorm(q,k) + RoPE, scatter q/k/v -> [NH][L][D] f16
// --------------------------------------------------------------------------
__global__ __launch_bounds__(128) void qkv_post_kernel(
    const _Float16* __restrict__ qkv, const float* __restrict__ pe,
    const float* __restrict__ qsc, const float* __restrict__ ksc,
    _Float16* __restrict__ Qo, _Float16* __restrict__ Ko, _Float16* __restrict__ Vo) {
  int l = blockIdx.x, h = blockIdx.y, d = threadIdx.x;
  const _Float16* base = qkv + (size_t)l * (3 * Hd) + h * Dh;
  float q = (float)base[d];
  float k = (float)base[Hd + d];
  float v = (float)base[2 * Hd + d];

  __shared__ float r1[4], r2[4];
  float a = q * q, b = k * k;
#pragma unroll
  for (int off = 16; off > 0; off >>= 1) {
    a += __shfl_down(a, off, 32);
    b += __shfl_down(b, off, 32);
  }
  int lane = d & 31, w = d >> 5;
  if (lane == 0) { r1[w] = a; r2[w] = b; }
  __syncthreads();
  float sq = r1[0] + r1[1] + r1[2] + r1[3];
  float sk = r2[0] + r2[1] + r2[2] + r2[3];
  float qn = q * rsqrtf(sq * (1.f / Dh) + 1e-6f) * qsc[d];
  float kn = k * rsqrtf(sk * (1.f / Dh) + 1e-6f) * ksc[d];

  __shared__ float sQn[128], sKn[128];
  sQn[d] = qn; sKn[d] = kn;
  __syncthreads();
  int i = d >> 1, r = d & 1;
  const float* p = pe + (((size_t)l * (Dh / 2) + i) * 2 + r) * 2;
  float qo = p[0] * sQn[i * 2] + p[1] * sQn[i * 2 + 1];
  float ko = p[0] * sKn[i * 2] + p[1] * sKn[i * 2 + 1];

  size_t o = ((size_t)h * Lq + l) * Dh + d;
  Qo[o] = (_Float16)qo;
  Ko[o] = (_Float16)ko;
  Vo[o] = (_Float16)v;
}

// --------------------------------------------------------------------------
// TDM staging of a contiguous 64x128-half K tile into a 136-half-pitch LDS
// tile: 1-row tensor of 8192 f16 elements, pad 4 dwords after every 64 dwords
// (reproduces the +8 half row pad).  Issued by one wave; EXEC is ignored.
// --------------------------------------------------------------------------
#if HAS_TDM
__device__ inline void tdm_load_k_tile(const _Float16* gsrc, _Float16* ldsDst) {
  unsigned lds_off = (unsigned)(unsigned long long)ldsDst;  // flat low bits == LDS offset
  unsigned long long ga = (unsigned long long)gsrc;
  u32x4 g0;
  g0[0] = 1u;                                            // count=1, user descriptor
  g0[1] = lds_off;                                       // lds_addr
  g0[2] = (unsigned)(ga & 0xffffffffu);                  // global_addr[31:0]
  g0[3] = (unsigned)((ga >> 32) & 0x1ffffffu) | (2u << 30);  // addr[56:32], type=2
  i32x8 g1;
  g1[0] = (1 << 16) | (1 << 20) | (5 << 22) | (3 << 25); // 2B elems, pad 4dw/64dw
  g1[1] = (int)(8192u << 16);                            // tensor_dim0 = 8192
  g1[2] = (int)(1u << 16);                               // tensor_dim1 = 1
  g1[3] = (int)(8192u << 16);                            // tile_dim0 = 8192
  g1[4] = 0;                                             // tile_dim1/2 unused
  g1[5] = 8192;                                          // tensor_dim0_stride
  g1[6] = 0;
  g1[7] = 0;
  i32x4 z4 = {0, 0, 0, 0};
#if __clang_major__ >= 23
  i32x8 z8 = {0, 0, 0, 0, 0, 0, 0, 0};
  __builtin_amdgcn_tensor_load_to_lds(g0, g1, z4, z4, z8, 0);
#else
  __builtin_amdgcn_tensor_load_to_lds(g0, g1, z4, z4, 0);
#endif
}
#endif

// --------------------------------------------------------------------------
// K5: flash attention, one block per (64-query tile, head); 8 waves.
// K tile staged by the Tensor Data Mover when available.
// --------------------------------------------------------------------------
__global__ __launch_bounds__(256) void attn_kernel(
    const _Float16* __restrict__ Q, const _Float16* __restrict__ Kc,
    const _Float16* __restrict__ V, _Float16* __restrict__ big_a) {
  __shared__ _Float16 sQ[64 * 136];
  __shared__ _Float16 sK[64 * 136];
  __shared__ _Float16 sVT[128 * 72];   // [d][key]
  __shared__ _Float16 sSP[64 * 72];    // S then P, f16
  __shared__ float sM[64], sL[64], sAl[64];

  int h = blockIdx.y;
  int q0 = blockIdx.x * 64;
  int tid = threadIdx.x, lane = tid & 31, w = tid >> 5;
  int wq = (w & 3) * 16;
  int wc = (w >> 2);        // 0 or 1
  int hi8 = (lane >> 4) << 3;
  int cn  = lane & 15;
  const float qscale = 0.08838834764831845f;  // 128^-0.5

  const _Float16* Qbase = Q + ((size_t)h * Lq + q0) * Dh;
#pragma unroll
  for (int t = 0; t < 4; ++t) {
    int chunk = tid + t * 256;
    int r = chunk >> 4, c0 = (chunk & 15) * 8;
    v8h val = *(const v8h*)(Qbase + (size_t)r * Dh + c0);
#pragma unroll
    for (int i = 0; i < 8; ++i) val[i] = (_Float16)((float)val[i] * qscale);
    *(v8h*)(sQ + r * 136 + c0) = val;
  }
  if (tid < 64) { sM[tid] = -1e30f; sL[tid] = 0.f; }
  v8f o[4] = {};
  __syncthreads();

  for (int kt = 0; kt < Lq; kt += 64) {
    const _Float16* Kbase = Kc + ((size_t)h * Lq + kt) * Dh;
    const _Float16* Vbase = V  + ((size_t)h * Lq + kt) * Dh;
#if HAS_TDM
    if (w == 0) tdm_load_k_tile(Kbase, sK);
#endif
#pragma unroll
    for (int t = 0; t < 4; ++t) {
      int chunk = tid + t * 256;
      int r = chunk >> 4, c0 = (chunk & 15) * 8;
#if !HAS_TDM
      *(v8h*)(sK + r * 136 + c0) = *(const v8h*)(Kbase + (size_t)r * Dh + c0);
#endif
      v8h vv = *(const v8h*)(Vbase + (size_t)r * Dh + c0);
#pragma unroll
      for (int i = 0; i < 8; ++i) sVT[(c0 + i) * 72 + r] = vv[i];
    }
#if HAS_TDM
    if (w == 0) wait_tensor_all();
#endif
    __syncthreads();

    // S = Q K^T  (16x32 strip per wave)
    v8f s0 = {}, s1 = {};
#pragma unroll
    for (int ks = 0; ks < Dh; ks += 32) {
      v16h af = frag16x32(sQ + wq * 136, 136, ks, lane);
      v16h b0 = frag16x32(sK + (wc * 32) * 136, 136, ks, lane);
      v16h b1 = frag16x32(sK + (wc * 32 + 16) * 136, 136, ks, lane);
      s0 = WMMA_F16(af, b0, s0);
      s1 = WMMA_F16(af, b1, s1);
    }
#pragma unroll
    for (int i = 0; i < 8; ++i) {
      int r = wq + i + hi8;
      sSP[r * 72 + wc * 32 + cn]      = (_Float16)s0[i];
      sSP[r * 72 + wc * 32 + 16 + cn] = (_Float16)s1[i];
    }
    __syncthreads();

    // online softmax (threads 0..63 each own one query row)
    if (tid < 64) {
      _Float16* srow = sSP + tid * 72;
      float mold = sM[tid], mx = mold;
      for (int c = 0; c < 64; ++c) mx = fmaxf(mx, (float)srow[c]);
      float al = __expf(mold - mx);
      float sum = 0.f;
      for (int c = 0; c < 64; ++c) {
        float pp = __expf((float)srow[c] - mx);
        srow[c] = (_Float16)pp;
        sum += pp;
      }
      sM[tid] = mx;
      sL[tid] = sL[tid] * al + sum;
      sAl[tid] = al;
    }
    __syncthreads();

    // rescale O by alpha(row) and accumulate P @ V
    float alv[8];
#pragma unroll
    for (int i = 0; i < 8; ++i) alv[i] = sAl[wq + i + hi8];
#pragma unroll
    for (int j = 0; j < 4; ++j)
#pragma unroll
      for (int i = 0; i < 8; ++i) o[j][i] *= alv[i];
#pragma unroll
    for (int ks = 0; ks < 64; ks += 32) {
      v16h af = frag16x32(sSP + wq * 72, 72, ks, lane);
#pragma unroll
      for (int j = 0; j < 4; ++j) {
        v16h bf = frag16x32(sVT + (wc * 64 + j * 16) * 72, 72, ks, lane);
        o[j] = WMMA_F16(af, bf, o[j]);
      }
    }
    __syncthreads();
  }

  float linv[8];
#pragma unroll
  for (int i = 0; i < 8; ++i) linv[i] = 1.f / sL[wq + i + hi8];
#pragma unroll
  for (int j = 0; j < 4; ++j) {
#pragma unroll
    for (int i = 0; i < 8; ++i) {
      int row = q0 + wq + i + hi8;
      int col = h * Dh + wc * 64 + j * 16 + cn;
      big_a[(size_t)row * BIGK + col] = (_Float16)(o[j][i] * linv[i]);
    }
  }
}

// --------------------------------------------------------------------------
// Launch
// --------------------------------------------------------------------------
extern "C" void kernel_launch(void* const* d_in, const int* in_sizes, int n_in,
                              void* d_out, int out_size, void* d_ws, size_t ws_size,
                              hipStream_t stream) {
  const float* x     = (const float*)d_in[0];
  const float* vec   = (const float*)d_in[1];
  const float* pe    = (const float*)d_in[2];
  const float* mod_w = (const float*)d_in[3];
  const float* mod_b = (const float*)d_in[4];
  const float* w1    = (const float*)d_in[5];
  const float* b1    = (const float*)d_in[6];
  const float* w2    = (const float*)d_in[7];
  const float* b2    = (const float*)d_in[8];
  const float* qsc   = (const float*)d_in[9];
  const float* ksc   = (const float*)d_in[10];
  float* out = (float*)d_out;

  char* ws = (char*)d_ws;
  size_t off = 0;
  float* sv = (float*)(ws + off);          off += 16384;                       // 3072 f32
  float* mod = (float*)(ws + off);         off += 65536;                       // 9216 f32
  _Float16* xmod = (_Float16*)(ws + off);  off += (size_t)Lq * Hd * 2;         // 12.6 MB
  _Float16* qkv  = (_Float16*)(ws + off);  off += (size_t)Lq * 3 * Hd * 2;     // 37.7 MB
  _Float16* Qb   = (_Float16*)(ws + off);  off += (size_t)NH * Lq * Dh * 2;    // 12.6 MB
  _Float16* Kb   = (_Float16*)(ws + off);  off += (size_t)NH * Lq * Dh * 2;    // 12.6 MB
  _Float16* Vb   = (_Float16*)(ws + off);  off += (size_t)NH * Lq * Dh * 2;    // 12.6 MB
  _Float16* bigA = (_Float16*)(ws + off);  off += (size_t)Lq * BIGK * 2;       // 62.9 MB

  // 1) silu(vec)
  silu_vec_kernel<<<(Hd + 255) / 256, 256, 0, stream>>>(vec, sv);
  // 2) mod vector (shift/scale/gate)
  mod_gemv_kernel<<<(3 * Hd) / 256, 256, 0, stream>>>(sv, mod_w, mod_b, mod);
  // 3) LayerNorm + modulation -> x_mod (f16)
  ln_mod_kernel<<<Lq, 256, 0, stream>>>(x, mod, xmod);
  // 4) QKV GEMM: [2048,3072] @ [9216,3072]^T
  wmma_gemm_kernel<EP_QKV><<<dim3((3 * Hd) / 64, Lq / 128), 256, 0, stream>>>(
      xmod, Hd, w1, Hd, b1, Hd, qkv, 3 * Hd, 0, nullptr, nullptr, nullptr);
  // 5) RMSNorm + RoPE + scatter q/k/v
  qkv_post_kernel<<<dim3(Lq, NH), 128, 0, stream>>>(qkv, pe, qsc, ksc, Qb, Kb, Vb);
  // 6) flash attention -> big_a[:, 0:3072]
  attn_kernel<<<dim3(Lq / 64, NH), 256, 0, stream>>>(Qb, Kb, Vb, bigA);
  // 7) fc1 + GELU: [2048,3072] @ [12288,3072]^T -> big_a[:, 3072:15360]
  wmma_gemm_kernel<EP_GELU><<<dim3(Mm / 64, Lq / 128), 256, 0, stream>>>(
      xmod, Hd, w1 + (size_t)(3 * Hd) * Hd, Hd, b1 + 3 * Hd, Hd,
      bigA, BIGK, Hd, nullptr, nullptr, nullptr);
  // 8) fused output GEMM (o-proj + fc2), residual + gate:
  //    out = x + gate * (big_a @ w2^T + 2*b2)
  wmma_gemm_kernel<EP_OUT><<<dim3(Hd / 64, Lq / 128), 256, 0, stream>>>(
      bigA, BIGK, w2, BIGK, b2, BIGK, nullptr, Hd, 0, out, x, mod);

  (void)in_sizes; (void)n_in; (void)out_size; (void)ws_size; (void)off;
}
